// AxisNet_52862457479510
// MI455X (gfx1250) — matrix operand order repaired and checked by reading, not compiled, not run on hardware
//
#include <hip/hip_runtime.h>
#include <hip/hip_bf16.h>
#include <math.h>

// ---------------------------------------------------------------------------
// AxisNet on MI455X (gfx1250): fp16 activations + v_wmma_f32_16x16x32_f16,
// BN folded into next-layer A-load as packed-f16 (v_pk_fma_f16 / v_pk_max),
// stats fused into the GEMM epilogue via LDS float atomics,
// ping-pong 128MB fp16 buffers.
// ---------------------------------------------------------------------------

typedef _Float16 f16;
typedef __attribute__((ext_vector_type(16))) _Float16 v16h;
typedef __attribute__((ext_vector_type(8)))  _Float16 v8h;
typedef __attribute__((ext_vector_type(8)))  float    v8f;

static constexpr int B_   = 8;
static constexpr int N_   = 4096;
static constexpr int K_   = 32;
static constexpr int POS  = B_ * K_ * N_;      // 1,048,576 conv positions
static constexpr int ROWS = B_ * N_;           // 32,768 FC rows
static constexpr int SROW = 64;                // fp16 channel stride (128B rows)

// ---- workspace layout (bytes). Assumes ws_size >= ~285 MB. ----------------
static constexpr size_t WS_P0   = 0;
static constexpr size_t WS_P1   = WS_P0 + (size_t)POS * SROW * 2;   // 128 MB
static constexpr size_t WS_H    = WS_P1 + (size_t)POS * SROW * 2;   // 128 MB
static constexpr size_t WS_F1   = WS_H  + (size_t)ROWS * SROW * 2;  // 4 MB
static constexpr size_t WS_F2   = WS_F1 + (size_t)ROWS * SROW * 2;
static constexpr size_t WS_F3   = WS_F2 + (size_t)ROWS * SROW * 2;
static constexpr size_t WS_STAT = WS_F3 + (size_t)ROWS * SROW * 2;  // 12*128 f32
static constexpr size_t WS_ITX  = WS_STAT + 12 * 128 * sizeof(float); // 13*64 float4

// ===========================================================================
// init: zero BN stats, zero all input-transform tables, seed identity entries
// (xyz channels of conv L0/L3/L6 consumers; FC0 consumes already-normalized h)
// itx entry = (scale, shift, relu_flag, unused)
// ===========================================================================
__global__ void k_init(float* __restrict__ stats, float4* __restrict__ itx)
{
    int t = blockIdx.x * blockDim.x + threadIdx.x;
    int stride = gridDim.x * blockDim.x;
    for (int i = t; i < 12 * 128; i += stride) stats[i] = 0.f;
    for (int i = t; i < 13 * 64; i += stride) {
        int tab = i >> 6, c = i & 63;
        float4 v = make_float4(0.f, 0.f, 0.f, 0.f);
        bool xyz_id = (tab == 0 || tab == 3 || tab == 6) && (c < 3); // raw xyz
        bool fc0_id = (tab == 9);                                    // pooled h
        if (xyz_id || fc0_id) v = make_float4(1.f, 0.f, 0.f, 0.f);   // no relu
        itx[i] = v;
    }
}

// ===========================================================================
// gather: xyz4[b,c,k,n] = xyz[b, neighbors[b,n,k], c] -> fp16 ch 0..2 of both
// ping-pong buffers (so stage concats find xyz in place).
// ===========================================================================
__global__ void k_gather(const float* __restrict__ xyz, const int* __restrict__ nbr,
                         f16* __restrict__ P0, f16* __restrict__ P1)
{
    int t = blockIdx.x * blockDim.x + threadIdx.x;   // pos = b*K*N + k*N + n
    if (t >= POS) return;
    int b = t / (K_ * N_);
    int r = t - b * K_ * N_;
    int k = r / N_;
    int n = r - k * N_;
    int idx = nbr[((size_t)b * N_ + n) * K_ + k];
    const float* s = xyz + ((size_t)b * N_ + idx) * 3;
    f16 x0 = (f16)s[0], x1 = (f16)s[1], x2 = (f16)s[2];
    size_t base = (size_t)t * SROW;
    P0[base + 0] = x0; P0[base + 1] = x1; P0[base + 2] = x2;
    P1[base + 0] = x0; P1[base + 1] = x1; P1[base + 2] = x2;
}

// ===========================================================================
// WMMA GEMM: Y[pos, out_off+o] = bias[o] + sum_c Wpad[o,c]*max(s_c*A[pos,c]+t_c, bd_c)
//  - A fp16 [pos][64]; BN+ReLU applied as packed-f16 fma+max with per-channel
//    lower bound bd (0 => relu, -65504 => identity, padding has s=t=0)
//  - weights zero-padded on the fly (padding channels contribute exactly 0)
//  - per-channel sum/sumsq for the *next* BN accumulated via LDS atomics
//  - KT = #32-wide K steps, NT = #16-wide output tiles (compile-time unroll)
// ===========================================================================
template <int KT, int NT>
__global__ __launch_bounds__(256)
void k_gemm(const f16* __restrict__ A, f16* __restrict__ Y,
            const float* __restrict__ W, const float* __restrict__ bias,
            const float4* __restrict__ itx, float* __restrict__ stats,
            int P, int in_c, int in_off, int out_c, int out_off)
{
    __shared__ __attribute__((aligned(16))) f16 s_sc[64];
    __shared__ __attribute__((aligned(16))) f16 s_sh[64];
    __shared__ __attribute__((aligned(16))) f16 s_bd[64];
    __shared__ float s_stat[128];
    int tid = threadIdx.x;
    if (tid < 64) {
        float4 v = itx[tid];
        s_sc[tid] = (f16)v.x;
        s_sh[tid] = (f16)v.y;
        s_bd[tid] = (v.z != 0.f) ? (f16)0.f : (f16)(-65504.f);
    }
    if (tid < 128) s_stat[tid] = 0.f;
    __syncthreads();

    int lane  = tid & 31;
    int wave  = blockIdx.x * 8 + (tid >> 5);
    int nWave = gridDim.x * 8;
    int nTile = P >> 4;

    int n  = lane & 15;        // output column / A row within tile
    int kg = lane >> 4;        // half-K group select (wave32 A/B layout)
    int koffBase = kg * 8;     // lane's K phase within a 32-wide step

    // --- hoist per-lane BN transform vectors out of the tile loop ----------
    v8h sc[KT][2], sh[KT][2], bd[KT][2];
#pragma unroll
    for (int kk = 0; kk < KT; ++kk) {
        int koff = kk * 32 + koffBase;
        sc[kk][0] = *(const v8h*)(s_sc + koff);
        sh[kk][0] = *(const v8h*)(s_sh + koff);
        bd[kk][0] = *(const v8h*)(s_bd + koff);
        sc[kk][1] = *(const v8h*)(s_sc + koff + 16);
        sh[kk][1] = *(const v8h*)(s_sh + koff + 16);
        bd[kk][1] = *(const v8h*)(s_bd + koff + 16);
    }

    // --- build B fragments once per wave (weights are tiny, L2-resident) ---
    v16h bfr[KT][NT];
#pragma unroll
    for (int kk = 0; kk < KT; ++kk)
#pragma unroll
        for (int nt = 0; nt < NT; ++nt) {
            int o = nt * 16 + n;
            v16h bf;
#pragma unroll
            for (int j = 0; j < 16; ++j) {
                int cw = kk * 32 + kg * 8 + (j & 7) + ((j >> 3) << 4) - in_off;   // weight channel
                float w = (o < out_c && cw >= 0 && cw < in_c) ? W[o * in_c + cw] : 0.f;
                bf[j] = (f16)w;
            }
            bfr[kk][nt] = bf;
        }

    // --- hoist bias out of the tile loop -----------------------------------
    float bo[NT];
#pragma unroll
    for (int nt = 0; nt < NT; ++nt) {
        int o = nt * 16 + n;
        bo[nt] = (o < out_c) ? bias[o] : 0.f;
    }

    float lsum[NT], lsq[NT];
#pragma unroll
    for (int nt = 0; nt < NT; ++nt) { lsum[nt] = 0.f; lsq[nt] = 0.f; }

    for (int tile = wave; tile < nTile; tile += nWave) {
        size_t rowBase = (size_t)tile << 4;
        const f16* ap = A + (rowBase + n) * SROW;

        // prefetch next grid-stride tile into caches (global_prefetch_b8)
        size_t nxt = rowBase + (size_t)nWave * 16;
        if (nxt < (size_t)P) __builtin_prefetch(A + (nxt + n) * SROW, 0, 1);

        v8f acc[NT];
#pragma unroll
        for (int nt = 0; nt < NT; ++nt) acc[nt] = (v8f){0.f,0.f,0.f,0.f,0.f,0.f,0.f,0.f};

#pragma unroll
        for (int kk = 0; kk < KT; ++kk) {
            int koff = kk * 32 + koffBase;
            v8h lo = *(const v8h*)(ap + koff);        // ch koff..koff+7
            v8h hi = *(const v8h*)(ap + koff + 16);   // ch koff+16..koff+23
            // packed BN+ReLU: v_pk_fma_f16 + v_pk_max_num_f16
            v8h t0 = __builtin_elementwise_max((v8h)(lo * sc[kk][0] + sh[kk][0]), bd[kk][0]);
            v8h t1 = __builtin_elementwise_max((v8h)(hi * sc[kk][1] + sh[kk][1]), bd[kk][1]);
            v16h a = __builtin_shufflevector(t0, t1, 0,1,2,3,4,5,6,7,
                                                     8,9,10,11,12,13,14,15);
#pragma unroll
            for (int nt = 0; nt < NT; ++nt)
                acc[nt] = __builtin_amdgcn_wmma_f32_16x16x32_f16(
                    false, a, false, bfr[kk][nt], (short)0, acc[nt], false, false);
        }

        // epilogue: add conv bias, store raw fp16 y, accumulate BN stats
#pragma unroll
        for (int nt = 0; nt < NT; ++nt) {
            int o = nt * 16 + n;
            if (o < out_c) {
                f16* yp = Y + (rowBase + (size_t)kg * 8) * SROW + out_off + o;
#pragma unroll
                for (int r = 0; r < 8; ++r) {        // C layout: M = kg*8 + r
                    float v = acc[nt][r] + bo[nt];
                    yp[(size_t)r * SROW] = (f16)v;
                    lsum[nt] += v;
                    lsq[nt]  += v * v;
                }
            }
        }
    }

    if (stats) {
#pragma unroll
        for (int nt = 0; nt < NT; ++nt) {
            int o = nt * 16 + n;
            if (o < out_c) {
                atomicAdd(&s_stat[o],      lsum[nt]);   // ds_add_f32
                atomicAdd(&s_stat[64 + o], lsq[nt]);
            }
        }
        __syncthreads();
        if (tid < 128) atomicAdd(&stats[tid], s_stat[tid]); // global_atomic_add_f32
    }
}

// ===========================================================================
// finalize BN: (sum,sumsq) -> per-channel (scale, shift, relu) for consumer
// ===========================================================================
__global__ void k_finalize(const float* __restrict__ stats,
                           const float* __restrict__ g, const float* __restrict__ beta,
                           int nch, float invP, float4* __restrict__ itx_dst, int off)
{
    int c = threadIdx.x;
    if (c < nch) {
        float mean = stats[c] * invP;
        float var  = stats[64 + c] * invP - mean * mean;
        float s = g[c] * rsqrtf(var + 1e-5f);
        float t = beta[c] - mean * s;
        itx_dst[off + c] = make_float4(s, t, 1.f, 0.f);
    }
}

// ===========================================================================
// maxpool over K with stage-final BN+ReLU applied on the fly
// h[b*N+n, o] = max_k relu(s_o * y[b,o,k,n] + t_o)
// ===========================================================================
__global__ void k_maxpool(const f16* __restrict__ Z, const float4* __restrict__ st,
                          f16* __restrict__ H)
{
    int t = blockIdx.x * blockDim.x + threadIdx.x;   // t = (b*N+n)*64 + o
    if (t >= ROWS * 64) return;
    int o   = t & 63;
    int row = t >> 6;
    int b   = row / N_;
    int n   = row - b * N_;
    float4 s = st[o];
    float m = -3.0e38f;
    size_t base = ((size_t)b * K_ * N_ + n) * SROW + o;
    for (int k = 0; k < K_; ++k) {
        float v = (float)Z[base + (size_t)k * N_ * SROW];
        v = fmaxf(fmaf(v, s.x, s.y), 0.f);
        m = fmaxf(m, v);
    }
    H[(size_t)row * SROW + o] = (f16)m;
}

// ===========================================================================
// axes epilogue: Gram-Schmidt frame from the 6 FC outputs per point
// ===========================================================================
__global__ void k_axes(const f16* __restrict__ F, float* __restrict__ out)
{
    int r = blockIdx.x * blockDim.x + threadIdx.x;
    if (r >= ROWS) return;
    const f16* p = F + (size_t)r * SROW;
    float a1x = (float)p[0], a1y = (float)p[1], a1z = (float)p[2];
    float a2x = (float)p[3], a2y = (float)p[4], a2z = (float)p[5];
    float n1 = sqrtf(a1x*a1x + a1y*a1y + a1z*a1z) + 1e-9f;
    float kf = (a1x*a2x + a1y*a2y + a1z*a2z) / (n1 * n1);
    float bx = a2x - kf*a1x, by = a2y - kf*a1y, bz = a2z - kf*a1z;
    float nb = sqrtf(bx*bx + by*by + bz*bz) + 1e-9f;
    float xx = bx/nb, xy = by/nb, xz = bz/nb;
    float zx = a1x/n1, zy = a1y/n1, zz = a1z/n1;
    float yx = zy*xz - zz*xy, yy = zz*xx - zx*xz, yz = zx*xy - zy*xx;
    out[(size_t)r*3 + 0] = xx; out[(size_t)r*3 + 1] = xy; out[(size_t)r*3 + 2] = xz;
    float* oy = out + (size_t)ROWS * 3;
    oy[(size_t)r*3 + 0] = yx; oy[(size_t)r*3 + 1] = yy; oy[(size_t)r*3 + 2] = yz;
    float* oz = out + (size_t)ROWS * 6;
    oz[(size_t)r*3 + 0] = zx; oz[(size_t)r*3 + 1] = zy; oz[(size_t)r*3 + 2] = zz;
}

// ===========================================================================
extern "C" void kernel_launch(void* const* d_in, const int* in_sizes, int n_in,
                              void* d_out, int out_size, void* d_ws, size_t ws_size,
                              hipStream_t stream)
{
    (void)in_sizes; (void)n_in; (void)out_size; (void)ws_size;
    const float* xyz = (const float*)d_in[0];
    const int*   nbr = (const int*)d_in[1];
    const float *cW[9], *cB[9], *bG[9], *bB[9], *fW[3], *fB[3], *fG[2], *fBt[2];
    for (int i = 0; i < 9; ++i) { cW[i] = (const float*)d_in[2 + i];
                                  cB[i] = (const float*)d_in[11 + i];
                                  bG[i] = (const float*)d_in[20 + i];
                                  bB[i] = (const float*)d_in[29 + i]; }
    for (int i = 0; i < 3; ++i) { fW[i] = (const float*)d_in[38 + i];
                                  fB[i] = (const float*)d_in[41 + i]; }
    for (int i = 0; i < 2; ++i) { fG[i]  = (const float*)d_in[44 + i];
                                  fBt[i] = (const float*)d_in[46 + i]; }

    char* ws = (char*)d_ws;
    f16* P0 = (f16*)(ws + WS_P0);
    f16* P1 = (f16*)(ws + WS_P1);
    f16* H  = (f16*)(ws + WS_H);
    f16* F1 = (f16*)(ws + WS_F1);
    f16* F2 = (f16*)(ws + WS_F2);
    f16* F3 = (f16*)(ws + WS_F3);
    float*  ST  = (float*)(ws + WS_STAT);
    float4* IT  = (float4*)(ws + WS_ITX);
#define STAT(i) (ST + (i) * 128)
#define ITX(i)  (IT + (i) * 64)

    const float invPc = 1.f / (float)POS;
    const float invPf = 1.f / (float)ROWS;
    const int GB = 1024;   // conv GEMM blocks (8192 waves, 8 tiles each)
    const int FB = 256;    // FC GEMM blocks

    k_init<<<4, 256, 0, stream>>>(ST, IT);
    k_gather<<<POS / 256, 256, 0, stream>>>(xyz, nbr, P0, P1);

    // ---- stage 1: 3 -> 6 -> 16 -> 16 (xyz at ch0..2, feats at ch3..) ----
    k_gemm<1,1><<<GB,256,0,stream>>>(P0,P1,cW[0],cB[0],ITX(0),STAT(0),POS, 3,0, 6,3);
    k_finalize<<<1,64,0,stream>>>(STAT(0),bG[0],bB[0], 6,invPc,ITX(1),3);
    k_gemm<1,1><<<GB,256,0,stream>>>(P1,P0,cW[1],cB[1],ITX(1),STAT(1),POS, 6,3,16,3);
    k_finalize<<<1,64,0,stream>>>(STAT(1),bG[1],bB[1],16,invPc,ITX(2),3);
    k_gemm<1,1><<<GB,256,0,stream>>>(P0,P1,cW[2],cB[2],ITX(2),STAT(2),POS,16,3,16,3);
    k_finalize<<<1,64,0,stream>>>(STAT(2),bG[2],bB[2],16,invPc,ITX(3),3);

    // ---- stage 2: concat(3+16)=19 -> 16 -> 16 -> 32 ----
    k_gemm<1,1><<<GB,256,0,stream>>>(P1,P0,cW[3],cB[3],ITX(3),STAT(3),POS,19,0,16,3);
    k_finalize<<<1,64,0,stream>>>(STAT(3),bG[3],bB[3],16,invPc,ITX(4),3);
    k_gemm<1,1><<<GB,256,0,stream>>>(P0,P1,cW[4],cB[4],ITX(4),STAT(4),POS,16,3,16,3);
    k_finalize<<<1,64,0,stream>>>(STAT(4),bG[4],bB[4],16,invPc,ITX(5),3);
    k_gemm<1,2><<<GB,256,0,stream>>>(P1,P0,cW[5],cB[5],ITX(5),STAT(5),POS,16,3,32,3);
    k_finalize<<<1,64,0,stream>>>(STAT(5),bG[5],bB[5],32,invPc,ITX(6),3);

    // ---- stage 3: concat(3+32)=35 -> 32 -> 32 -> 64 ----
    k_gemm<2,2><<<GB,256,0,stream>>>(P0,P1,cW[6],cB[6],ITX(6),STAT(6),POS,35,0,32,3);
    k_finalize<<<1,64,0,stream>>>(STAT(6),bG[6],bB[6],32,invPc,ITX(7),3);
    k_gemm<2,2><<<GB,256,0,stream>>>(P1,P0,cW[7],cB[7],ITX(7),STAT(7),POS,32,3,32,3);
    k_finalize<<<1,64,0,stream>>>(STAT(7),bG[7],bB[7],32,invPc,ITX(8),3);
    k_gemm<2,4><<<GB,256,0,stream>>>(P0,P1,cW[8],cB[8],ITX(8),STAT(8),POS,32,3,64,0);
    k_finalize<<<1,64,0,stream>>>(STAT(8),bG[8],bB[8],64,invPc,ITX(12),0);

    // ---- maxpool over K with stage-final BN+ReLU fused ----
    k_maxpool<<<(ROWS * 64) / 256, 256, 0, stream>>>(P1, ITX(12), H);

    // ---- FC head: 64 -> 32 -> 32 -> 6 (BN over rows after FC0/FC1) ----
    k_gemm<2,2><<<FB,256,0,stream>>>(H ,F1,fW[0],fB[0],ITX(9), STAT(9), ROWS,64,0,32,0);
    k_finalize<<<1,64,0,stream>>>(STAT(9), fG[0],fBt[0],32,invPf,ITX(10),0);
    k_gemm<1,2><<<FB,256,0,stream>>>(F1,F2,fW[1],fB[1],ITX(10),STAT(10),ROWS,32,0,32,0);
    k_finalize<<<1,64,0,stream>>>(STAT(10),fG[1],fBt[1],32,invPf,ITX(11),0);
    k_gemm<1,1><<<FB,256,0,stream>>>(F2,F3,fW[2],fB[2],ITX(11),nullptr, ROWS,32,0, 6,0);

    // ---- orthonormal axes ----
    k_axes<<<ROWS / 256, 256, 0, stream>>>(F3, (float*)d_out);
}
